// MultiQueryAttention_20650202759983
// MI455X (gfx1250) — matrix operand order repaired and checked by reading, not compiled, not run on hardware
//
#include <hip/hip_runtime.h>
#include <hip/hip_bf16.h>

#define D_MODEL 1024
#define NUM_HEADS 16
#define HEAD_DIM 64
#define SEQ 2048
#define BATCH 2

typedef __attribute__((__ext_vector_type__(16))) __bf16 v16bf;
typedef __attribute__((__ext_vector_type__(8)))  float  v8f;
typedef __attribute__((__ext_vector_type__(4)))  unsigned int u32x4;

union FragB16 {
    v16bf v;
    __bf16 h[16];
    u32x4 q[2];
};

__device__ __forceinline__ v8f wmma_bf16(v16bf a, v16bf b, v8f c) {
    // 8 args: (neg_a, A, neg_b, B, c_mod, C, reuse_a, reuse_b)
    return __builtin_amdgcn_wmma_f32_16x16x32_bf16(false, a, false, b, (short)0, c,
                                                   false, false);
}

// xor-16 lane swap within wave32 (ds_swizzle group-of-32: xor=0x10, and=0x1f)
__device__ __forceinline__ float swz16(float x) {
    return __int_as_float(__builtin_amdgcn_ds_swizzle(__float_as_int(x), 0x401F));
}

// CDNA5 async memory->LDS DMA (16B per lane), tracked by ASYNCcnt.
__device__ __forceinline__ void async_load_b128(unsigned lds_off, const void* gptr) {
    asm volatile("global_load_async_to_lds_b128 %0, %1, off"
                 :
                 : "v"(lds_off), "v"(gptr)
                 : "memory");
}
__device__ __forceinline__ void wait_asynccnt0() {
    asm volatile("s_wait_asynccnt 0x0" ::: "memory");
}

__global__ void f32_to_bf16_kernel(const float* __restrict__ in,
                                   __bf16* __restrict__ out, int n) {
    for (int i = blockIdx.x * blockDim.x + threadIdx.x; i < n;
         i += gridDim.x * blockDim.x) {
        out[i] = (__bf16)in[i];
    }
}

// C[M,N] = A[M,K] @ W[K,N] + bias ; A,W bf16 row-major. BM=128 BN=64 BK=32.
// 256 threads = 8 waves arranged 4(M) x 2(N); each wave -> 32x32 = 4 WMMA tiles.
// Double-buffered LDS; A tile staged with global_load_async_to_lds_b128.
template <bool OUT_F32>
__global__ __launch_bounds__(256) void gemm_bf16_kernel(
    const __bf16* __restrict__ A, const __bf16* __restrict__ W,
    const float* __restrict__ bias, void* __restrict__ Cout, int M, int N, int Kd) {
    __shared__ __attribute__((aligned(16))) __bf16 a_lds[2][128 * 40];  // [m][k]
    __shared__ __attribute__((aligned(16))) __bf16 wt_lds[2][64 * 40];  // W^T [n][k]

    const int tid = threadIdx.x;
    const int wave = tid >> 5;
    const int lane = tid & 31;
    const int ln = lane & 15;
    const int half = lane >> 4;
    const int wm = wave & 3;   // M quadrant (rows wm*32)
    const int wn = wave >> 2;  // N half (cols wn*32)
    const int m0 = blockIdx.x * 128;
    const int n0 = blockIdx.y * 64;

    const int arow = tid >> 1, ach = (tid & 1) * 16;  // A stage: 128x32, 1 b128/thread
    const int wrow = tid >> 2, wch = (tid & 3) * 16;  // W stage: 32x64 (tid<128)
    union WReg { u32x4 q; __bf16 h[16]; };

    const v8f vzero = {0.f, 0.f, 0.f, 0.f, 0.f, 0.f, 0.f, 0.f};
    v8f c[2][2] = {{vzero, vzero}, {vzero, vzero}};

    // prologue: stage tile 0 into buffer 0
    async_load_b128((unsigned)(size_t)&a_lds[0][arow * 40 + ach],
                    A + (size_t)(m0 + arow) * Kd + ach);
    if (tid < 128) {
        WReg t;
        t.q = *(const u32x4*)(W + (size_t)wrow * N + n0 + wch);
#pragma unroll
        for (int i = 0; i < 16; ++i) wt_lds[0][(wch + i) * 40 + wrow] = t.h[i];
    }

    const int ntiles = Kd / 32;
    for (int it = 0; it < ntiles; ++it) {
        const int buf = it & 1;
        const int nbuf = buf ^ 1;
        wait_asynccnt0();
        __syncthreads();

        const bool have_next = (it + 1 < ntiles);
        WReg tw;
        tw.q = (u32x4){0u, 0u, 0u, 0u};
        if (have_next) {
            const int k0n = (it + 1) * 32;
            async_load_b128((unsigned)(size_t)&a_lds[nbuf][arow * 40 + ach],
                            A + (size_t)(m0 + arow) * Kd + k0n + ach);
            if (tid < 128)
                tw.q = *(const u32x4*)(W + (size_t)(k0n + wrow) * N + n0 + wch);
        }

        FragB16 af[2], bfr[2];
#pragma unroll
        for (int i = 0; i < 2; ++i) {
            // A-frag: lane m = l%16 ; k = half*16 + j (16 contiguous)
            const __bf16* ar = &a_lds[buf][(wm * 32 + i * 16 + ln) * 40 + half * 16];
            af[i].q[0] = *(const u32x4*)(ar);
            af[i].q[1] = *(const u32x4*)(ar + 8);
            // B-frag: lane n = l%16 ; k = half*16 + j (16 contiguous)
            const __bf16* wr = &wt_lds[buf][(wn * 32 + i * 16 + ln) * 40 + half * 16];
            bfr[i].q[0] = *(const u32x4*)(wr);
            bfr[i].q[1] = *(const u32x4*)(wr + 8);
        }
#pragma unroll
        for (int i = 0; i < 2; ++i)
#pragma unroll
            for (int j = 0; j < 2; ++j)
                c[i][j] = wmma_bf16(af[i].v, bfr[j].v, c[i][j]);

        if (have_next && tid < 128) {  // transpose-store next W tile (overlaps WMMA)
#pragma unroll
            for (int i = 0; i < 16; ++i) wt_lds[nbuf][(wch + i) * 40 + wrow] = tw.h[i];
        }
    }

#pragma unroll
    for (int i = 0; i < 2; ++i)
#pragma unroll
        for (int j = 0; j < 2; ++j) {
            int n = n0 + wn * 32 + j * 16 + ln;
            float bv = bias[n];
#pragma unroll
            for (int r = 0; r < 8; ++r) {
                int m = m0 + wm * 32 + i * 16 + r + 8 * half;
                float val = c[i][j][r] + bv;
                if (OUT_F32)
                    ((float*)Cout)[(size_t)m * N + n] = val;
                else
                    ((__bf16*)Cout)[(size_t)m * N + n] = (__bf16)val;
            }
        }
}

// Flash-style MQA attention. Block = 4 waves; each wave owns 16 queries.
// Computes S^T = K @ Q^T and O^T = V^T @ P^T so softmax stats are per-lane.
// K tile staged via async DMA; V^T via reg transpose; double-buffered LDS.
__global__ __launch_bounds__(128) void mqa_attn_kernel(
    const __bf16* __restrict__ Q, const __bf16* __restrict__ K,
    const __bf16* __restrict__ V, __bf16* __restrict__ Att) {
    const int QB = SEQ / 64;  // 32 query blocks per (b,h)
    int id = blockIdx.x;
    const int qblk = id % QB; id /= QB;
    const int h = id % NUM_HEADS; id /= NUM_HEADS;
    const int b = id;

    const int tid = threadIdx.x;
    const int wave = tid >> 5;
    const int lane = tid & 31;
    const int ln = lane & 15;
    const int half = lane >> 4;
    const int qbase = qblk * 64 + wave * 16;

    __shared__ __attribute__((aligned(16))) __bf16 k_lds[2][32 * 72];   // [key][hd]
    __shared__ __attribute__((aligned(16))) __bf16 vt_lds[2][64 * 40];  // [hd][key]

    // Q^T B-fragments (held in registers all kernel):
    // lane n = query = l%16 ; k_hd = step*32 + half*16 + j (16 contiguous bf16)
    FragB16 qf[2];
    const __bf16* qrow =
        Q + (size_t)(b * SEQ + qbase + ln) * D_MODEL + h * HEAD_DIM;
#pragma unroll
    for (int st = 0; st < 2; ++st) {
        qf[st].q[0] = *(const u32x4*)(qrow + st * 32 + half * 16);
        qf[st].q[1] = *(const u32x4*)(qrow + st * 32 + half * 16 + 8);
    }

    const v8f vzero = {0.f, 0.f, 0.f, 0.f, 0.f, 0.f, 0.f, 0.f};
    v8f o[4] = {vzero, vzero, vzero, vzero};  // O^T tiles over hd (4 x 16)
    float mrow = -3.0e38f;
    float lrow = 0.f;

    const __bf16* Kb = K + (size_t)b * SEQ * HEAD_DIM;
    const __bf16* Vb = V + (size_t)b * SEQ * HEAD_DIM;
    const int srow = tid >> 2, sch = (tid & 3) * 16;  // staging: 32 rows x 4 chunks
    union VReg { u32x4 q; __bf16 h[16]; };

    // prologue: stage tile 0 into buffer 0
    async_load_b128((unsigned)(size_t)&k_lds[0][srow * 72 + sch],
                    Kb + (size_t)srow * HEAD_DIM + sch);
    {
        VReg t;
        t.q = *(const u32x4*)(Vb + (size_t)srow * HEAD_DIM + sch);
#pragma unroll
        for (int i = 0; i < 16; ++i) vt_lds[0][(sch + i) * 40 + srow] = t.h[i];
    }

    for (int kb = 0; kb < SEQ; kb += 32) {
        const int buf = (kb >> 5) & 1;
        const int nbuf = buf ^ 1;
        wait_asynccnt0();
        __syncthreads();

        const bool have_next = (kb + 32 < SEQ);
        VReg tv;
        tv.q = (u32x4){0u, 0u, 0u, 0u};
        if (have_next) {
            async_load_b128((unsigned)(size_t)&k_lds[nbuf][srow * 72 + sch],
                            Kb + (size_t)(kb + 32 + srow) * HEAD_DIM + sch);
            tv.q = *(const u32x4*)(Vb + (size_t)(kb + 32 + srow) * HEAD_DIM + sch);
        }

        // scores^T : two 16(key)x16(query) C tiles, K-dim = hd = 64 (2 WMMA each)
        v8f sc[2];
#pragma unroll
        for (int t = 0; t < 2; ++t) {
            v8f acc = vzero;
#pragma unroll
            for (int st = 0; st < 2; ++st) {
                FragB16 ka;  // A = K: lane m = key = l%16 ; k_hd per A layout
                const __bf16* kr =
                    &k_lds[buf][(t * 16 + ln) * 72 + st * 32 + half * 16];
                ka.q[0] = *(const u32x4*)(kr);
                ka.q[1] = *(const u32x4*)(kr + 8);
                acc = wmma_bf16(ka.v, qf[st].v, acc);
            }
            sc[t] = acc;
        }

        // online softmax over keys (per-lane: query = l%16 for both halves)
        float mt = -3.0e38f;
#pragma unroll
        for (int t = 0; t < 2; ++t)
#pragma unroll
            for (int r = 0; r < 8; ++r) {
                sc[t][r] *= 0.125f;  // 1/sqrt(64)
                mt = fmaxf(mt, sc[t][r]);
            }
        mt = fmaxf(mt, swz16(mt));
        float mnew = fmaxf(mrow, mt);
        float alpha = __expf(mrow - mnew);
        float psum = 0.f;
#pragma unroll
        for (int t = 0; t < 2; ++t)
#pragma unroll
            for (int r = 0; r < 8; ++r) {
                float p = __expf(sc[t][r] - mnew);
                sc[t][r] = p;
                psum += p;
            }
        psum += swz16(psum);
        lrow = lrow * alpha + psum;
        mrow = mnew;
#pragma unroll
        for (int mt4 = 0; mt4 < 4; ++mt4)
#pragma unroll
            for (int r = 0; r < 8; ++r) o[mt4][r] *= alpha;

        // pack P^T B-fragment: lane n = query ; k(key) = half*16 + j
        // half==0 lane: j=0..7 <- own tile0 regs, j=8..15 <- partner tile0 regs
        // half==1 lane: j=0..7 <- partner tile1 regs, j=8..15 <- own tile1 regs
        FragB16 pf;
#pragma unroll
        for (int r = 0; r < 8; ++r) {
            float p0x = swz16(sc[0][r]);
            float p1x = swz16(sc[1][r]);
            float jlo = half ? p1x : sc[0][r];
            float jhi = half ? sc[1][r] : p0x;
            pf.h[r] = (__bf16)jlo;
            pf.h[r + 8] = (__bf16)jhi;
        }

        // O^T += V^T @ P^T  (4 tiles over hd)
#pragma unroll
        for (int mt4 = 0; mt4 < 4; ++mt4) {
            FragB16 va;  // A = V^T: lane m = hd = mt4*16 + l%16
            const __bf16* vr = &vt_lds[buf][(mt4 * 16 + ln) * 40 + half * 16];
            va.q[0] = *(const u32x4*)(vr);
            va.q[1] = *(const u32x4*)(vr + 8);
            o[mt4] = wmma_bf16(va.v, pf.v, o[mt4]);
        }

        if (have_next) {  // transpose-store next V tile (overlaps WMMA latency)
#pragma unroll
            for (int i = 0; i < 16; ++i) vt_lds[nbuf][(sch + i) * 40 + srow] = tv.h[i];
        }
    }

    const float inv = 1.0f / lrow;
    const int q = qbase + ln;
#pragma unroll
    for (int mt4 = 0; mt4 < 4; ++mt4)
#pragma unroll
        for (int r = 0; r < 8; ++r) {
            int hd = mt4 * 16 + r + 8 * half;
            Att[(size_t)(b * SEQ + q) * D_MODEL + h * HEAD_DIM + hd] =
                (__bf16)(o[mt4][r] * inv);
        }
}

extern "C" void kernel_launch(void* const* d_in, const int* in_sizes, int n_in,
                              void* d_out, int out_size, void* d_ws, size_t ws_size,
                              hipStream_t stream) {
    (void)in_sizes; (void)n_in; (void)out_size; (void)ws_size;
    const float* x  = (const float*)d_in[0];
    const float* Wq = (const float*)d_in[1];
    const float* bq = (const float*)d_in[2];
    const float* Wk = (const float*)d_in[3];
    const float* bk = (const float*)d_in[4];
    const float* Wv = (const float*)d_in[5];
    const float* bv = (const float*)d_in[6];
    const float* Wo = (const float*)d_in[7];
    const float* bo = (const float*)d_in[8];
    float* out = (float*)d_out;

    const int M = BATCH * SEQ;  // 4096
    char* ws = (char*)d_ws;
    size_t off = 0;
    auto alloc_bf16 = [&](size_t elems) {
        __bf16* p = (__bf16*)(ws + off);
        off += elems * sizeof(__bf16);
        return p;
    };
    __bf16* xb  = alloc_bf16((size_t)M * D_MODEL);
    __bf16* qb  = alloc_bf16((size_t)M * D_MODEL);
    __bf16* att = alloc_bf16((size_t)M * D_MODEL);
    __bf16* wqb = alloc_bf16((size_t)D_MODEL * D_MODEL);
    __bf16* wob = alloc_bf16((size_t)D_MODEL * D_MODEL);
    __bf16* wkb = alloc_bf16((size_t)D_MODEL * HEAD_DIM);
    __bf16* wvb = alloc_bf16((size_t)D_MODEL * HEAD_DIM);
    __bf16* kb  = alloc_bf16((size_t)M * HEAD_DIM);
    __bf16* vb  = alloc_bf16((size_t)M * HEAD_DIM);

    auto cvt = [&](const float* src, __bf16* dst, int n) {
        int blocks = (n + 255) / 256;
        if (blocks > 8192) blocks = 8192;
        f32_to_bf16_kernel<<<blocks, 256, 0, stream>>>(src, dst, n);
    };
    cvt(x,  xb,  M * D_MODEL);
    cvt(Wq, wqb, D_MODEL * D_MODEL);
    cvt(Wk, wkb, D_MODEL * HEAD_DIM);
    cvt(Wv, wvb, D_MODEL * HEAD_DIM);
    cvt(Wo, wob, D_MODEL * D_MODEL);

    // Q projection: [4096,1024] = xb @ Wq + bq (bf16 out)
    gemm_bf16_kernel<false><<<dim3(M / 128, D_MODEL / 64), 256, 0, stream>>>(
        xb, wqb, bq, qb, M, D_MODEL, D_MODEL);
    // K/V projections: [4096,64]
    gemm_bf16_kernel<false><<<dim3(M / 128, 1), 256, 0, stream>>>(
        xb, wkb, bk, kb, M, HEAD_DIM, D_MODEL);
    gemm_bf16_kernel<false><<<dim3(M / 128, 1), 256, 0, stream>>>(
        xb, wvb, bv, vb, M, HEAD_DIM, D_MODEL);

    // attention: grid over (b, h, 64-query blocks)
    mqa_attn_kernel<<<BATCH * NUM_HEADS * (SEQ / 64), 128, 0, stream>>>(
        qb, kb, vb, att);

    // output projection: out = att @ Wo + bo (f32 out)
    gemm_bf16_kernel<true><<<dim3(M / 128, D_MODEL / 64), 256, 0, stream>>>(
        att, wob, bo, out, M, D_MODEL, D_MODEL);
}
